// MLPMoE_69552700391642
// MI455X (gfx1250) — matrix operand order repaired and checked by reading, not compiled, not run on hardware
//
#include <hip/hip_runtime.h>

// ---------------- CDNA5 (gfx1250) MoE MLP with hard gumbel routing ----------
// B=4 S=2048 D=768 H=3072 E=8.  Forward of hard gumbel-softmax == pure argmax
// selection, so we route each token to exactly one expert and run one MLP.
// fp32 accuracy via f16 hi/lo split + 3x v_wmma_f32_16x16x32_f16 per K-slab.

typedef __attribute__((ext_vector_type(16))) _Float16 v16h;
typedef __attribute__((ext_vector_type(8)))  _Float16 v8h;
typedef __attribute__((ext_vector_type(8)))  float    v8f;

#define NTOK   8192
#define DMODEL 768
#define HDIM   3072
#define NEXP   8
#define MAXTILES 512              // ceil(8192/16)

// LDS layout for the MoE kernel (activations pre-split into f16 hi/lo)
#define XT_STRIDE 776             // f16 elems per x row (768 + pad), 16B-multiple rows
#define HT_STRIDE 3080            // f16 elems per h row (3072 + pad), 16B-multiple rows
#define WB_STRIDE 33              // f32: 32 lanes + pad
#define XT_HALFS  (16 * XT_STRIDE)            // per hi/lo array
#define HT_HALFS  (16 * HT_STRIDE)
#define WB_FLOATS (8 * 16 * WB_STRIDE)        // 8 waves x 16x33 f32
#define SMEM_BYTES (2 * XT_HALFS * 2 + 2 * HT_HALFS * 2 + WB_FLOATS * 4 + 64)

// ---------------------------------------------------------------------------
// k0: zero counters (cnt[8], offs[8], cursor[8])
__global__ void k0_init(int* ws) {
    int i = threadIdx.x;
    if (i < 24) ws[i] = 0;
}

// k1: gating.  One wave32 per token: scores = x.Wg + bg + gumbel, argmax.
__global__ __launch_bounds__(256) void k1_gate(const float* __restrict__ x,
                                               const float* __restrict__ gumbel,
                                               const float* __restrict__ Wg,
                                               const float* __restrict__ bg,
                                               int* __restrict__ cnt,
                                               int* __restrict__ expertOf) {
    int wv = threadIdx.x >> 5, lane = threadIdx.x & 31;
    int t = blockIdx.x * 8 + wv;
    float acc[NEXP];
#pragma unroll
    for (int e = 0; e < NEXP; ++e) acc[e] = 0.0f;
    for (int d = lane; d < DMODEL; d += 32) {
        float xv = x[(size_t)t * DMODEL + d];
        const float* wg = Wg + d * NEXP;
#pragma unroll
        for (int e = 0; e < NEXP; ++e) acc[e] += xv * wg[e];
    }
#pragma unroll
    for (int off = 16; off >= 1; off >>= 1) {
#pragma unroll
        for (int e = 0; e < NEXP; ++e) acc[e] += __shfl_xor(acc[e], off, 32);
    }
    if (lane == 0) {
        float best = -3.4e38f; int be = 0;
#pragma unroll
        for (int e = 0; e < NEXP; ++e) {
            float s = acc[e] + bg[e] + gumbel[(size_t)t * NEXP + e];
            if (s > best) { best = s; be = e; }   // first-max tie break, like argmax
        }
        expertOf[t] = be;
        atomicAdd(&cnt[be], 1);
    }
}

// k2: tiny exclusive scan of 8 counts
__global__ void k2_scan(int* ws) {
    // ws: [0..7]=cnt  [8..15]=offs  [16..23]=cursor
    if (threadIdx.x == 0) {
        int run = 0;
        for (int e = 0; e < NEXP; ++e) { ws[8 + e] = run; run += ws[e]; }
    }
}

// k3: scatter token ids into per-expert contiguous lists
__global__ __launch_bounds__(256) void k3_scatter(const int* __restrict__ expertOf,
                                                  const int* __restrict__ offs,
                                                  int* __restrict__ cursor,
                                                  int* __restrict__ list) {
    int t = blockIdx.x * blockDim.x + threadIdx.x;
    int e = expertOf[t];
    int pos = atomicAdd(&cursor[e], 1);
    list[offs[e] + pos] = t;
}

// ---------------------------------------------------------------------------
// fp32-accurate 16x16 C tile via 3x v_wmma_f32_16x16x32_f16 per K=32 slab.
// A comes pre-split as f16 hi/lo LDS arrays (row-major, aStride f16/row) ->
// A-side needs no conversions, only two 16B LDS vector loads per matrix.
// Wc: global f32, row-major, wStride floats per K row, 16 columns used;
// it is transposed through a per-wave LDS buffer and split on the fly.
__device__ __forceinline__ v8f mm_tile(const _Float16* __restrict__ Ahi,
                                       const _Float16* __restrict__ Alo, int aStride,
                                       const float* __restrict__ Wc, int wStride,
                                       int Ktot, float* wb) {
    const int lane = threadIdx.x & 31;
    const int hh = lane >> 4;        // half-wave: selects K sub-block
    const int mn = lane & 15;        // A row (M) / B column (N)
    v8f c;
#pragma unroll
    for (int i = 0; i < 8; ++i) c[i] = 0.0f;

    for (int kb = 0; kb < Ktot; kb += 32) {
        // --- stage 32 K-rows x 16 N-cols of W through LDS (transpose) ---
        const float* wrow = Wc + (size_t)(kb + lane) * wStride;
        if (kb + 32 < Ktot)
            __builtin_prefetch(wrow + (size_t)32 * wStride, 0, 3); // global_prefetch_b8
        float4 q0 = *(const float4*)(wrow + 0);
        float4 q1 = *(const float4*)(wrow + 4);
        float4 q2 = *(const float4*)(wrow + 8);
        float4 q3 = *(const float4*)(wrow + 12);
        wb[ 0*WB_STRIDE + lane] = q0.x; wb[ 1*WB_STRIDE + lane] = q0.y;
        wb[ 2*WB_STRIDE + lane] = q0.z; wb[ 3*WB_STRIDE + lane] = q0.w;
        wb[ 4*WB_STRIDE + lane] = q1.x; wb[ 5*WB_STRIDE + lane] = q1.y;
        wb[ 6*WB_STRIDE + lane] = q1.z; wb[ 7*WB_STRIDE + lane] = q1.w;
        wb[ 8*WB_STRIDE + lane] = q2.x; wb[ 9*WB_STRIDE + lane] = q2.y;
        wb[10*WB_STRIDE + lane] = q2.z; wb[11*WB_STRIDE + lane] = q2.w;
        wb[12*WB_STRIDE + lane] = q3.x; wb[13*WB_STRIDE + lane] = q3.y;
        wb[14*WB_STRIDE + lane] = q3.z; wb[15*WB_STRIDE + lane] = q3.w;
        // wave-internal ordering only (wb is wave-private); DS ops are in-order
        asm volatile("s_wait_dscnt 0" ::: "memory");

        // --- gather B in contiguous K order (same as A): lane holds col mn, K=(hh*16 .. +15) ---
        const float* bs = wb + mn * WB_STRIDE + hh * 16;
        v16h bhi, blo;
#pragma unroll
        for (int j = 0; j < 16; ++j) {
            float v = bs[j];
            _Float16 hv = (_Float16)v;
            bhi[j] = hv; blo[j] = (_Float16)(v - (float)hv);
        }
        // --- A in contiguous K order: row mn; elems 0..15 -> K=kb+16*hh+j ---
        const _Float16* ah = Ahi + mn * aStride + kb + 16 * hh;
        const _Float16* al = Alo + mn * aStride + kb + 16 * hh;
        v8h ah0 = *(const v8h*)(ah);
        v8h ah1 = *(const v8h*)(ah + 8);
        v8h al0 = *(const v8h*)(al);
        v8h al1 = *(const v8h*)(al + 8);
        v16h ahi = __builtin_shufflevector(ah0, ah1, 0,1,2,3,4,5,6,7,8,9,10,11,12,13,14,15);
        v16h alo = __builtin_shufflevector(al0, al1, 0,1,2,3,4,5,6,7,8,9,10,11,12,13,14,15);

        // emulated-fp32: (ah+al)(bh+bl) ~= ah*bh + al*bh + ah*bl
        c = __builtin_amdgcn_wmma_f32_16x16x32_f16(false, ahi, false, bhi, (short)0, c, false, false);
        c = __builtin_amdgcn_wmma_f32_16x16x32_f16(false, alo, false, bhi, (short)0, c, false, false);
        c = __builtin_amdgcn_wmma_f32_16x16x32_f16(false, ahi, false, blo, (short)0, c, false, false);
    }
    return c;
}

__device__ __forceinline__ float gelu_tanh(float v) {
    // jax.nn.gelu (approximate=True): 0.5v(1+tanh(z)) == v*sigmoid(2z), branch-free
    float z2 = 1.5957691216057308f * (v + 0.044715f * v * v * v);
    return v * __builtin_amdgcn_rcpf(1.0f + __expf(-z2));
}

// k4: one 16-token tile of one expert per block.  256 threads = 8 wave32s.
__global__ __launch_bounds__(256) void k4_moe(const float* __restrict__ x,
                                              const float* __restrict__ W1,
                                              const float* __restrict__ b1,
                                              const float* __restrict__ W2,
                                              const float* __restrict__ b2,
                                              float* __restrict__ out,
                                              const int* __restrict__ cnt,
                                              const int* __restrict__ offs,
                                              const int* __restrict__ list) {
    const int e = blockIdx.y;
    const int cntE = cnt[e];
    const int tile = blockIdx.x;
    if (tile * 16 >= cntE) return;
    const int valid = (cntE - tile * 16 < 16) ? (cntE - tile * 16) : 16;

    extern __shared__ float smem[];
    _Float16* xThi = (_Float16*)smem;                 // [16][XT_STRIDE]
    _Float16* xTlo = xThi + XT_HALFS;
    _Float16* hThi = xTlo + XT_HALFS;                 // [16][HT_STRIDE]
    _Float16* hTlo = hThi + HT_HALFS;
    float*    wbA  = (float*)(hTlo + HT_HALFS);       // 8 x [16][33] f32
    int*      tokL = (int*)(wbA + WB_FLOATS);         // 16 token ids

    const int tid = threadIdx.x;
    if (tid < 16) {
        int mi = tid < valid - 1 ? tid : valid - 1;   // clamp pad rows
        tokL[tid] = list[offs[e] + tile * 16 + mi];
    }
    __syncthreads();
    for (int i = tid; i < 16 * DMODEL; i += 256) {
        int m = i / DMODEL, d = i - m * DMODEL;
        float v = x[(size_t)tokL[m] * DMODEL + d];
        _Float16 hv = (_Float16)v;
        xThi[m * XT_STRIDE + d] = hv;
        xTlo[m * XT_STRIDE + d] = (_Float16)(v - (float)hv);
    }
    __syncthreads();

    const int lane = tid & 31, wv = tid >> 5;
    const int hh = lane >> 4, mn = lane & 15;
    float* wb = wbA + wv * (16 * WB_STRIDE);
    const float* W1e = W1 + (size_t)e * DMODEL * HDIM;
    const float* W2e = W2 + (size_t)e * HDIM * DMODEL;
    const float* b1e = b1 + e * HDIM;
    const float* b2e = b2 + e * DMODEL;

    // ---- stage 1: h = gelu(x @ W1 + b1), each wave owns 384 hidden cols ----
    for (int nt = 0; nt < 24; ++nt) {
        int nc0 = wv * 384 + nt * 16;
        v8f c = mm_tile(xThi, xTlo, XT_STRIDE, W1e + nc0, HDIM, DMODEL, wb);
        float bias = b1e[nc0 + mn];
#pragma unroll
        for (int i = 0; i < 8; ++i) {
            float g = gelu_tanh(c[i] + bias);          // C/D: M = i + 8*hh, N = mn
            _Float16 ghi = (_Float16)g;
            int idx = (i + 8 * hh) * HT_STRIDE + nc0 + mn;
            hThi[idx] = ghi;
            hTlo[idx] = (_Float16)(g - (float)ghi);
        }
    }
    __syncthreads();

    // ---- stage 2: out = h @ W2 + b2, each wave owns 96 output cols ----
    for (int nt = 0; nt < 6; ++nt) {
        int nc0 = wv * 96 + nt * 16;
        v8f c = mm_tile(hThi, hTlo, HT_STRIDE, W2e + nc0, DMODEL, HDIM, wb);
        float bias = b2e[nc0 + mn];
#pragma unroll
        for (int i = 0; i < 8; ++i) {
            int m = i + 8 * hh;
            if (tile * 16 + m < cntE)
                out[(size_t)tokL[m] * DMODEL + nc0 + mn] = c[i] + bias;
        }
    }
}

// ---------------------------------------------------------------------------
extern "C" void kernel_launch(void* const* d_in, const int* in_sizes, int n_in,
                              void* d_out, int out_size, void* d_ws, size_t ws_size,
                              hipStream_t stream) {
    (void)in_sizes; (void)n_in; (void)out_size; (void)ws_size;
    const float* x      = (const float*)d_in[0];
    const float* gumbel = (const float*)d_in[1];
    const float* Wg     = (const float*)d_in[2];
    const float* bg     = (const float*)d_in[3];
    const float* W1     = (const float*)d_in[4];
    const float* b1     = (const float*)d_in[5];
    const float* W2     = (const float*)d_in[6];
    const float* b2     = (const float*)d_in[7];
    float* out = (float*)d_out;

    int* ws       = (int*)d_ws;
    int* offs     = ws + 8;        // 8 (cnt is ws[0..7], cursor ws[16..23])
    int* cursor   = ws + 16;
    int* expertOf = ws + 24;       // 8192
    int* list     = ws + 24 + NTOK;// 8192

    (void)hipFuncSetAttribute(reinterpret_cast<const void*>(k4_moe),
                              hipFuncAttributeMaxDynamicSharedMemorySize, SMEM_BYTES);

    k0_init<<<1, 32, 0, stream>>>(ws);
    k1_gate<<<NTOK / 8, 256, 0, stream>>>(x, gumbel, Wg, bg, ws, expertOf);
    k2_scan<<<1, 1, 0, stream>>>(ws);
    k3_scatter<<<NTOK / 256, 256, 0, stream>>>(expertOf, offs, cursor, list);
    dim3 grid(MAXTILES, NEXP);
    k4_moe<<<grid, 256, SMEM_BYTES, stream>>>(x, W1, b1, W2, b2, out, ws, offs, list);
}
